// UnivNetLVCResidualBlock_29540785062323
// MI455X (gfx1250) — matrix operand order repaired and checked by reading, not compiled, not run on hardware
//
#include <hip/hip_runtime.h>

typedef __attribute__((ext_vector_type(16))) _Float16 v16h;
typedef __attribute__((ext_vector_type(8)))  _Float16 v8h;
typedef __attribute__((ext_vector_type(8)))  float    v8f;

#define BB   16
#define CC   64
#define LL   16384
#define HOP  256
#define KSZ  3
#define DILA 3
#define KLN  64

// LDS geometry (time-major, f16, rows padded to 144B => every 16B-slot aligned)
#define XT   288   // x tile extent: global t = t0 - 12 + tt
#define GT   272   // g tile extent: global t = t0 - 8  + tg
#define ROWH 72    // halves per row (64 data + 8 pad) = 144 bytes

__device__ __forceinline__ float lrelu(float v) { return v > 0.f ? v : 0.2f * v; }

__global__ __launch_bounds__(256, 1)
void univnet_lvc_block(const float* __restrict__ x,
                       const float* __restrict__ kern,   // (B, C, 2C, KS, KL)
                       const float* __restrict__ bias,   // (B, 2C, KL)
                       const float* __restrict__ cw,     // (C, C, KS)
                       const float* __restrict__ cb,     // (C,)
                       float* __restrict__ out)          // (B, C, L)
{
  __shared__ __align__(16) _Float16 xs[XT * ROWH];   // lrelu(x), f16, halo ±12
  __shared__ __align__(16) _Float16 gs[GT * ROWH];   // conv output, f16, halo ±8
  __shared__ __align__(16) float    xr[HOP * CC];    // raw x for the residual

  const int tid  = threadIdx.x;
  const int lane = tid & 31;
  const int wave = tid >> 5;
  const int lh   = lane & 15;   // lane within half-wave
  const int hi   = lane >> 4;   // which half of the wave

  const int kl = blockIdx.x;    // conditioning frame
  const int b  = blockIdx.y;    // batch
  const int t0 = kl * HOP;      // frame start in global time

  const int m2 = wave & 3;      // LVC: channel tile inside each GLU half
  const int nh = wave >> 2;     // LVC: which half of the 16 N tiles

  // ---------------- Phase 0: stage x into LDS ------------------------------
  // lrelu(x) as f16 (time-major) for the conv GEMM, raw fp32 for the residual.
  {
    const bool interior = (t0 >= 12) && (t0 + (XT - 12) <= LL);
    for (int idx = tid; idx < CC * (XT / 4); idx += 256) {
      const int  i   = idx / (XT / 4);
      const int  tt4 = (idx - i * (XT / 4)) * 4;
      const long t   = (long)t0 + tt4 - 12;
      float4 v;
      if (interior) {
        v = *(const float4*)&x[((size_t)(b * CC + i)) * LL + t];
      } else {
        const float* row = &x[((size_t)(b * CC + i)) * LL];
        v.x = (t + 0 >= 0 && t + 0 < LL) ? row[t + 0] : 0.f;
        v.y = (t + 1 >= 0 && t + 1 < LL) ? row[t + 1] : 0.f;
        v.z = (t + 2 >= 0 && t + 2 < LL) ? row[t + 2] : 0.f;
        v.w = (t + 3 >= 0 && t + 3 < LL) ? row[t + 3] : 0.f;
      }
      xs[(tt4 + 0) * ROWH + i] = (_Float16)lrelu(v.x);
      xs[(tt4 + 1) * ROWH + i] = (_Float16)lrelu(v.y);
      xs[(tt4 + 2) * ROWH + i] = (_Float16)lrelu(v.z);
      xs[(tt4 + 3) * ROWH + i] = (_Float16)lrelu(v.w);
      if (tt4 >= 12 && tt4 < 268) {             // tl = tt-12 in [0, HOP)
        xr[(tt4 - 12 + 0) * CC + i] = v.x;
        xr[(tt4 - 12 + 1) * CC + i] = v.y;
        xr[(tt4 - 12 + 2) * CC + i] = v.z;
        xr[(tt4 - 12 + 3) * CC + i] = v.w;
      }
    }
  }

  // ---------------- LVC A-fragment + bias gather (issued early) ------------
  // Per-(b,kl) kernel weights: pure gather, each value used exactly once on
  // the whole device -> non-temporal loads; latency overlaps the conv phase.
  v16h AS[6], AT[6];
  float bS[8], bT[8];
  {
    for (int s = 0; s < 6; ++s) {
      const int oS = m2 * 16 + lh;
      for (int e = 0; e < 16; ++e) {
        const int K = 32 * s + 8 * hi + ((e < 8) ? e : e + 8);
        const int k = K >> 6, i = K & 63;
        const size_t base = (size_t)(b * CC + i) * (2 * CC);
        AS[s][e] = (_Float16)__builtin_nontemporal_load(
            &kern[((base + oS)      * KSZ + k) * KLN + kl]);
        AT[s][e] = (_Float16)__builtin_nontemporal_load(
            &kern[((base + oS + 64) * KSZ + k) * KLN + kl]);
      }
    }
    for (int r = 0; r < 8; ++r) {
      const int c = m2 * 16 + 8 * hi + r;
      bS[r] = bias[((size_t)b * 2 * CC + c)      * KLN + kl];
      bT[r] = bias[((size_t)b * 2 * CC + c + 64) * KLN + kl];
    }
  }
  __syncthreads();

  // ---------------- Phase 1: dilated conv GEMM  M=64, N=GT, K=192 ----------
  // K ordering: K = k*64 + i  (tap-major, channel-minor)
  {
    const int m = wave & 3;     // out-channel tile (4 x 16 = 64)
    const int p = wave >> 2;    // N-tile parity (2 waves per M tile)

    // A fragments (16x32 f16 lane layout, ISA 7.12.2); conv_w is tiny/cached.
    v16h A[6];
    for (int s = 0; s < 6; ++s) {
      const int o = m * 16 + lh;
      for (int e = 0; e < 16; ++e) {
        const int K = 32 * s + 8 * hi + ((e < 8) ? e : e + 8);
        const int k = K >> 6, i = K & 63;
        A[s][e] = (_Float16)cw[(o * CC + i) * KSZ + k];
      }
    }

    for (int nt = p; nt < GT / 16; nt += 2) {
      const int tg = nt * 16 + lh;            // this lane's output column
      v8f acc = {};
      for (int s = 0; s < 6; ++s) {
        const int k  = s >> 1;
        const int i0 = (s & 1) * 32 + 8 * hi;
        const int tt = tg + 1 + 3 * k;        // (tg-8) + (k-1)*DILA + 12
        v16h Bf;
        *((v8h*)&Bf)     = *(const v8h*)(&xs[tt * ROWH + i0]);
        *((v8h*)&Bf + 1) = *(const v8h*)(&xs[tt * ROWH + i0 + 16]);
        acc = __builtin_amdgcn_wmma_f32_16x16x32_f16(false, A[s], false, Bf,
                                                     (short)0, acc, false, false);
      }
      // epilogue: + conv_b, lrelu, zero outside [0,L) (matches _lvc zero-pad)
      v8h pk;
      const long t = (long)t0 + tg - 8;
      for (int r = 0; r < 8; ++r) {
        const int o = m * 16 + 8 * hi + r;
        float h = acc[r] + cb[o];
        h = lrelu(h);
        if (t < 0 || t >= LL) h = 0.f;
        pk[r] = (_Float16)h;
      }
      *(v8h*)(&gs[tg * ROWH + m * 16 + 8 * hi]) = pk;
    }
  }
  __syncthreads();

  // ---------------- Phase 2: LVC GEMM M=128, N=256, K=192 + GLU + residual -
  {
    const int c0 = m2 * 16 + 8 * hi;          // first of this lane's 8 channels

    for (int nt = nh * 8; nt < nh * 8 + 8; ++nt) {
      const int tl = nt * 16 + lh;            // frame-local time
      v8f accS = {}, accT = {};
      for (int s = 0; s < 6; ++s) {
        const int k  = s >> 1;
        const int i0 = (s & 1) * 32 + 8 * hi;
        const int tg = tl + 7 + k;            // g column for tap k (pad = 1)
        v16h Bf;
        *((v8h*)&Bf)     = *(const v8h*)(&gs[tg * ROWH + i0]);
        *((v8h*)&Bf + 1) = *(const v8h*)(&gs[tg * ROWH + i0 + 16]);
        accS = __builtin_amdgcn_wmma_f32_16x16x32_f16(false, AS[s], false, Bf,
                                                      (short)0, accS, false, false);
        accT = __builtin_amdgcn_wmma_f32_16x16x32_f16(false, AT[s], false, Bf,
                                                      (short)0, accT, false, false);
      }
      // residual from LDS (raw fp32), GLU gate, non-temporal write-once store
      const float4 r0 = *(const float4*)&xr[tl * CC + c0];
      const float4 r1 = *(const float4*)&xr[tl * CC + c0 + 4];
      const float res[8] = { r0.x, r0.y, r0.z, r0.w, r1.x, r1.y, r1.z, r1.w };
      for (int r = 0; r < 8; ++r) {
        const int   c    = c0 + r;
        const float hs   = accS[r] + bS[r];
        const float ht   = accT[r] + bT[r];
        const float gate = (1.f / (1.f + __expf(-hs))) * tanhf(ht);
        const size_t gidx = ((size_t)(b * CC + c)) * LL + (size_t)(t0 + tl);
        __builtin_nontemporal_store(res[r] + gate, &out[gidx]);
      }
    }
  }
}

extern "C" void kernel_launch(void* const* d_in, const int* in_sizes, int n_in,
                              void* d_out, int out_size, void* d_ws, size_t ws_size,
                              hipStream_t stream)
{
  const float* x    = (const float*)d_in[0];  // hidden_states (B,C,L)
  const float* kern = (const float*)d_in[1];  // kernel (B,C,2C,KS,KL)
  const float* bias = (const float*)d_in[2];  // bias (B,2C,KL)
  const float* cw   = (const float*)d_in[3];  // conv_w (C,C,KS)
  const float* cb   = (const float*)d_in[4];  // conv_b (C,)
  float* out = (float*)d_out;

  dim3 grid(KLN, BB);   // one workgroup per (frame, batch)
  univnet_lvc_block<<<grid, 256, 0, stream>>>(x, kern, bias, cw, cb, out);
}
